// MixtureOfExperts_85847806312745
// MI455X (gfx1250) — matrix operand order repaired and checked by reading, not compiled, not run on hardware
//
#include <hip/hip_runtime.h>

// ---------------------------------------------------------------------------
// MoE forward for MI455X (gfx1250, wave32, WMMA bf16 16x16x32, fp32 accum).
// Sparse top-2 routing; double-buffered LDS + register-pipelined staging;
// TDM (tensor_load_to_lds) async DMA for expert-GEMM2 A tiles.
// ---------------------------------------------------------------------------

typedef __bf16 bf16;
typedef __attribute__((ext_vector_type(16))) __bf16 v16bf;
typedef __attribute__((ext_vector_type(8)))  __bf16 v8bf;
typedef __attribute__((ext_vector_type(4)))  __bf16 v4bf;
typedef __attribute__((ext_vector_type(8)))  float  v8f;
typedef unsigned int u32x4 __attribute__((ext_vector_type(4)));
typedef int i32x4 __attribute__((ext_vector_type(4)));
typedef int i32x8 __attribute__((ext_vector_type(8)));

#define B_TOK  16384
#define TEXT_D 768
#define IMG_D  1024
#define HID    1024
#define OUTD   1024
#define NEXP   8
#define COMB   2048
#define NSLOT  (B_TOK * 2)      // exactly 2 routed rows per token

// GEMM tiling: workgroup 128x128x32, 8 waves (2M x 4N), wave tile 64x32.
#define BM 128
#define BN 128
#define BK 32
// LDS row stride 40 bf16 = 80 B: 16B-aligned segments; start bank 20*r mod 64
// enumerates all 16 four-bank groups over 16 rows -> conflict-free per half.
#define LSTR 40

#if __has_builtin(__builtin_amdgcn_tensor_load_to_lds)
#define HAVE_TDM 1
#else
#define HAVE_TDM 0
#endif

// ---- WMMA fragment loads from LDS (layouts per CDNA5 ISA 7.12.2) ----------
static __device__ __forceinline__ v16bf frag_a(const bf16* As, int m_base) {
  int l = threadIdx.x & 31;
  const bf16* p = As + (m_base + (l & 15)) * LSTR + ((l >> 4) << 3); // K half
  v8bf lo = *(const v8bf*)(p);        // K = khalf + 0..7   (VGPR 0-3)
  v8bf hi = *(const v8bf*)(p + 16);   // K = khalf + 16..23 (VGPR 4-7)
  v16bf r;
#pragma unroll
  for (int i = 0; i < 8; ++i) { r[i] = lo[i]; r[i + 8] = hi[i]; }
  return r;
}

static __device__ __forceinline__ v16bf frag_b(const bf16* Bs, int n_base) {
  // Bs stored transposed [n][k]; lane l: N = n_base+(l&15), K base (l>>4)*8.
  int l = threadIdx.x & 31;
  const bf16* p = Bs + (n_base + (l & 15)) * LSTR + ((l >> 4) << 3);
  v8bf lo = *(const v8bf*)(p);        // K = kbase + 0..7
  v8bf hi = *(const v8bf*)(p + 16);   // K = kbase + 16..23
  v16bf r;
#pragma unroll
  for (int i = 0; i < 8; ++i) { r[i] = lo[i]; r[i + 8] = hi[i]; }
  return r;
}

static __device__ __forceinline__ void mma_step(const bf16* As, const bf16* Bs,
                                                int m_wave, int n_wave,
                                                v8f (&acc)[4][2]) {
  v16bf b0 = frag_b(Bs, n_wave);
  v16bf b1 = frag_b(Bs, n_wave + 16);
#pragma unroll
  for (int mt = 0; mt < 4; ++mt) {
    v16bf a = frag_a(As, m_wave + mt * 16);
    acc[mt][0] = __builtin_amdgcn_wmma_f32_16x16x32_bf16(
        false, a, false, b0, (short)0, acc[mt][0], false, false);
    acc[mt][1] = __builtin_amdgcn_wmma_f32_16x16x32_bf16(
        false, a, false, b1, (short)0, acc[mt][1], false, false);
  }
}

// ---- register-staged tiles (load global -> regs; later regs -> LDS) -------
struct ARegF { float4 v[4]; };                // 128x32 fp32 A tile
static __device__ __forceinline__ void a_f32_load(ARegF& r, const float* A,
                                                  int lda, int m0, int k0) {
  int t = threadIdx.x, row = t >> 3, c = (t & 7) << 2;
#pragma unroll
  for (int p = 0; p < 4; ++p)
    r.v[p] = *(const float4*)(A + (size_t)(m0 + row + 32 * p) * lda + k0 + c);
}
static __device__ __forceinline__ void a_f32_store(bf16* As, const ARegF& r) {
  int t = threadIdx.x, row = t >> 3, c = (t & 7) << 2;
#pragma unroll
  for (int p = 0; p < 4; ++p) {
    v4bf b; b[0] = (bf16)r.v[p].x; b[1] = (bf16)r.v[p].y;
    b[2] = (bf16)r.v[p].z; b[3] = (bf16)r.v[p].w;
    *(v4bf*)(As + (row + 32 * p) * LSTR + c) = b;
  }
}

struct ARegB { v8bf v[2]; };                  // 128x32 bf16 A tile (gathered)
static __device__ __forceinline__ void a_bf16_load(ARegB& r, const bf16* src,
                                                   int lds_, const int* tok,
                                                   int row0, int k0) {
  int t = threadIdx.x, row = t >> 2, s = (t & 3) << 3;
#pragma unroll
  for (int p = 0; p < 2; ++p) {
    int gi = row0 + row + 64 * p;
    gi = gi < (NSLOT - 1) ? gi : (NSLOT - 1);       // clamp tail
    int tk = tok ? tok[gi] : gi;
    r.v[p] = *(const v8bf*)(src + (size_t)tk * lds_ + k0 + s);
  }
}
static __device__ __forceinline__ void a_bf16_store(bf16* As, const ARegB& r) {
  int t = threadIdx.x, row = t >> 2, s = (t & 3) << 3;
#pragma unroll
  for (int p = 0; p < 2; ++p)
    *(v8bf*)(As + (row + 64 * p) * LSTR + s) = r.v[p];
}

struct BRegF { float4 v[4]; };                // 32x128 fp32 W tile
static __device__ __forceinline__ void b_f32_load(BRegF& r, const float* W,
                                                  int ldw, int k0, int n0) {
  int t = threadIdx.x, kk = t >> 5, n4 = (t & 31) << 2;
#pragma unroll
  for (int p = 0; p < 4; ++p)
    r.v[p] = *(const float4*)(W + (size_t)(k0 + kk + 8 * p) * ldw + n0 + n4);
}
static __device__ __forceinline__ void b_f32_store(bf16* Bs, const BRegF& r) {
  int t = threadIdx.x, kk = t >> 5, n4 = (t & 31) << 2;
#pragma unroll
  for (int p = 0; p < 4; ++p) {
    int k = kk + 8 * p;
    Bs[(n4 + 0) * LSTR + k] = (bf16)r.v[p].x;       // transpose: [n][k]
    Bs[(n4 + 1) * LSTR + k] = (bf16)r.v[p].y;
    Bs[(n4 + 2) * LSTR + k] = (bf16)r.v[p].z;
    Bs[(n4 + 3) * LSTR + k] = (bf16)r.v[p].w;
  }
}

// ---- Tensor Data Mover: async bf16 tile (32 wide x `rows` tall) -> LDS ----
#if HAVE_TDM
static __device__ __forceinline__ void tdm_load_tile_bf16(
    unsigned lds_off, const bf16* gptr, int rows, int row_stride_elems) {
  unsigned long long ga = (unsigned long long)(uintptr_t)gptr;
  u32x4 g0;
  g0[0] = 1u;                                        // count=1, user mode
  g0[1] = lds_off;                                   // lds_addr (bytes)
  g0[2] = (unsigned)(ga & 0xFFFFFFFFu);              // global_addr[31:0]
  g0[3] = (unsigned)((ga >> 32) & 0x1FFFFFFu) | (2u << 30);  // addr hi | type=2
  i32x8 g1;
  // data_size=1(2B) | pad_enable | pad_interval=3 (16 DW = one 64B row)
  // | pad_amount=3 (4 DW = 16 B) -> LDS row pitch 80 B == LSTR.
  g1[0] = (1 << 16) | (1 << 20) | (3 << 22) | (3 << 25);
  g1[1] = (32u << 16);                               // tensor_dim0 = 32
  g1[2] = ((unsigned)rows << 16);                    // dim0 hi=0 | tensor_dim1 lo
  g1[3] = (32u << 16);                               // tensor_dim1 hi=0 | tile_dim0=32
  g1[4] = (unsigned)rows & 0xFFFFu;                  // tile_dim1=rows, tile_dim2=0
  g1[5] = (unsigned)row_stride_elems;                // tensor_dim0_stride lo
  g1[6] = 0;
  g1[7] = 0;
  i32x4 gz4 = {0, 0, 0, 0};                          // groups 2/3: 2D tile
  i32x8 gz8 = {0, 0, 0, 0, 0, 0, 0, 0};
  __builtin_amdgcn_tensor_load_to_lds(g0, g1, gz4, gz4, gz8, 0);
}
#endif

// ---------------------------------------------------------------------------
// 1) projections: combined[b, 0:1024]=text@Wt+bt, [1024:2048]=image@Wi+bi
// ---------------------------------------------------------------------------
__global__ __launch_bounds__(256) void proj_kernel(
    const float* __restrict__ text, const float* __restrict__ image,
    const float* __restrict__ Wt, const float* __restrict__ bt,
    const float* __restrict__ Wi, const float* __restrict__ bi,
    bf16* __restrict__ combined) {
  __shared__ __align__(16) bf16 As[2][BM * LSTR];
  __shared__ __align__(16) bf16 Bs[2][BN * LSTR];
  int m0 = blockIdx.x * BM;
  int ny = blockIdx.y;                       // 0..15
  bool is_text = ny < 8;
  const float* A    = is_text ? text : image;
  const float* W    = is_text ? Wt : Wi;
  const float* bias = is_text ? bt : bi;
  int K  = is_text ? TEXT_D : IMG_D;
  int n0 = (ny & 7) * BN;
  int wave = threadIdx.x >> 5;
  int m_wave = (wave >> 2) * 64;
  int n_wave = (wave & 3) * 32;
  v8f acc[4][2] = {};
  ARegF ar; BRegF br;
  a_f32_load(ar, A, K, m0, 0);
  b_f32_load(br, W, HID, 0, n0);
  a_f32_store(As[0], ar);
  b_f32_store(Bs[0], br);
  __syncthreads();
  int nk = K / BK;
  for (int i = 0; i < nk; ++i) {
    int cur = i & 1;
    if (i + 1 < nk) {                         // prefetch next tile into regs
      a_f32_load(ar, A, K, m0, (i + 1) * BK);
      b_f32_load(br, W, HID, (i + 1) * BK, n0);
    }
    mma_step(As[cur], Bs[cur], m_wave, n_wave, acc);
    if (i + 1 < nk) {
      a_f32_store(As[cur ^ 1], ar);
      b_f32_store(Bs[cur ^ 1], br);
    }
    __syncthreads();
  }
  int l = threadIdx.x & 31;
  int mofs = (l >> 4) * 8;
#pragma unroll
  for (int nt = 0; nt < 2; ++nt) {
    int ncol = n0 + n_wave + nt * 16 + (l & 15);
    float bv = bias[ncol];
    int gcol = is_text ? ncol : (HID + ncol);
#pragma unroll
    for (int mt = 0; mt < 4; ++mt)
#pragma unroll
      for (int r = 0; r < 8; ++r) {
        int row = m0 + m_wave + mt * 16 + mofs + r;
        combined[(size_t)row * COMB + gcol] = (bf16)(acc[mt][nt][r] + bv);
      }
  }
}

// ---------------------------------------------------------------------------
// 2) gating: wave-per-token dot(combined, Wg)+bg+noise, top-2, softmax
// ---------------------------------------------------------------------------
__global__ __launch_bounds__(256) void gate_kernel(
    const bf16* __restrict__ combined, const float* __restrict__ Wg,
    const float* __restrict__ bg, const float* __restrict__ noise,
    int* __restrict__ choice_e, float* __restrict__ choice_g,
    int* __restrict__ counts) {
  __shared__ bf16 sWg[COMB * NEXP];                     // 32 KB
  for (int i = threadIdx.x; i < COMB * NEXP; i += 256) sWg[i] = (bf16)Wg[i];
  __syncthreads();
  int wave = threadIdx.x >> 5, l = threadIdx.x & 31;
  int t = blockIdx.x * 8 + wave;
  const bf16* row = combined + (size_t)t * COMB;
  float acc[NEXP] = {};
  for (int d = l; d < COMB; d += 32) {
    float c = (float)row[d];
#pragma unroll
    for (int e = 0; e < NEXP; ++e) acc[e] += c * (float)sWg[d * NEXP + e];
  }
#pragma unroll
  for (int e = 0; e < NEXP; ++e)
#pragma unroll
    for (int off = 16; off; off >>= 1) acc[e] += __shfl_xor(acc[e], off, 32);
  if (l == 0) {
    float nl[NEXP];
#pragma unroll
    for (int e = 0; e < NEXP; ++e) nl[e] = acc[e] + bg[e] + noise[t * NEXP + e];
    int e0 = 0; float v0 = nl[0];
#pragma unroll
    for (int e = 1; e < NEXP; ++e) if (nl[e] > v0) { v0 = nl[e]; e0 = e; }
    int e1 = (e0 == 0) ? 1 : 0; float v1 = nl[e1];
#pragma unroll
    for (int e = 0; e < NEXP; ++e)
      if (e != e0 && nl[e] > v1) { v1 = nl[e]; e1 = e; }
    float m = fmaxf(v0, v1);
    float x0 = __expf(v0 - m), x1 = __expf(v1 - m);
    float inv = 1.0f / (x0 + x1);
    choice_e[t * 2 + 0] = e0; choice_g[t * 2 + 0] = x0 * inv;
    choice_e[t * 2 + 1] = e1; choice_g[t * 2 + 1] = x1 * inv;
    atomicAdd(&counts[e0], 1);
    atomicAdd(&counts[e1], 1);
  }
}

__global__ void init_kernel(int* counts, int* cursors) {
  int t = threadIdx.x;
  if (t < NEXP) { counts[t] = 0; cursors[t] = 0; }
}

__global__ void scan_kernel(const int* counts, int* bases) {
  if (threadIdx.x == 0) {
    int run = 0;
    for (int e = 0; e < NEXP; ++e) { bases[e] = run; run += counts[e]; }
  }
}

__global__ __launch_bounds__(256) void build_kernel(
    const int* __restrict__ choice_e, const float* __restrict__ choice_g,
    const int* __restrict__ bases, int* __restrict__ cursors,
    int* __restrict__ tok_of_row, int* __restrict__ slot_of_row,
    float* __restrict__ g_of_row) {
  int s = blockIdx.x * 256 + threadIdx.x;      // slot id = token*2 + k
  if (s >= NSLOT) return;
  int e = choice_e[s];
  int pos = atomicAdd(&cursors[e], 1);         // order-free: rows independent
  int row = bases[e] + pos;
  tok_of_row[row]  = s >> 1;
  slot_of_row[row] = s;
  g_of_row[row]    = choice_g[s];
}

// ---------------------------------------------------------------------------
// 3) expert GEMM1: h = relu(gather(combined) @ W1[e] + b1[e])
// ---------------------------------------------------------------------------
__global__ __launch_bounds__(256) void expert1_kernel(
    const bf16* __restrict__ combined, const float* __restrict__ W1,
    const float* __restrict__ b1, const int* __restrict__ counts,
    const int* __restrict__ bases, const int* __restrict__ tok_of_row,
    bf16* __restrict__ h) {
  __shared__ __align__(16) bf16 As[2][BM * LSTR];
  __shared__ __align__(16) bf16 Bs[2][BN * LSTR];
  int e = blockIdx.z;
  int cnt = counts[e];
  int mloc0 = blockIdx.x * BM;
  if (mloc0 >= cnt) return;
  int base = bases[e];
  int n0 = blockIdx.y * BN;
  const float* W = W1 + (size_t)e * COMB * HID;
  int wave = threadIdx.x >> 5;
  int m_wave = (wave >> 2) * 64;
  int n_wave = (wave & 3) * 32;
  v8f acc[4][2] = {};
  ARegB ar; BRegF br;
  a_bf16_load(ar, combined, COMB, tok_of_row, base + mloc0, 0);
  b_f32_load(br, W, HID, 0, n0);
  a_bf16_store(As[0], ar);
  b_f32_store(Bs[0], br);
  __syncthreads();
  int nk = COMB / BK;
  for (int i = 0; i < nk; ++i) {
    int cur = i & 1;
    if (i + 1 < nk) {
      a_bf16_load(ar, combined, COMB, tok_of_row, base + mloc0, (i + 1) * BK);
      b_f32_load(br, W, HID, (i + 1) * BK, n0);
    }
    mma_step(As[cur], Bs[cur], m_wave, n_wave, acc);
    if (i + 1 < nk) {
      a_bf16_store(As[cur ^ 1], ar);
      b_f32_store(Bs[cur ^ 1], br);
    }
    __syncthreads();
  }
  int l = threadIdx.x & 31;
  int mofs = (l >> 4) * 8;
#pragma unroll
  for (int nt = 0; nt < 2; ++nt) {
    int ncol = n0 + n_wave + nt * 16 + (l & 15);
    float bv = b1[e * HID + ncol];
#pragma unroll
    for (int mt = 0; mt < 4; ++mt)
#pragma unroll
      for (int r = 0; r < 8; ++r) {
        int mloc = mloc0 + m_wave + mt * 16 + mofs + r;
        if (mloc < cnt)
          h[(size_t)(base + mloc) * HID + ncol] =
              (bf16)fmaxf(acc[mt][nt][r] + bv, 0.0f);
      }
  }
}

// ---------------------------------------------------------------------------
// 4) expert GEMM2: slot_out[slot] = g * (h @ W2[e] + b2[e])
//    A tiles (contiguous bf16 rows of h) DMA'd by the Tensor Data Mover.
// ---------------------------------------------------------------------------
__global__ __launch_bounds__(256) void expert2_kernel(
    const bf16* __restrict__ h, const float* __restrict__ W2,
    const float* __restrict__ b2, const int* __restrict__ counts,
    const int* __restrict__ bases, const int* __restrict__ slot_of_row,
    const float* __restrict__ g_of_row, bf16* __restrict__ slot_out) {
  __shared__ __align__(16) bf16 As[2][BM * LSTR];
  __shared__ __align__(16) bf16 Bs[2][BN * LSTR];
  int e = blockIdx.z;
  int cnt = counts[e];
  int mloc0 = blockIdx.x * BM;
  if (mloc0 >= cnt) return;
  int base = bases[e];
  int n0 = blockIdx.y * BN;
  const float* W = W2 + (size_t)e * HID * OUTD;
  int wave = threadIdx.x >> 5;
  int m_wave = (wave >> 2) * 64;
  int n_wave = (wave & 3) * 32;
  v8f acc[4][2] = {};
  int rows = NSLOT - (base + mloc0); if (rows > BM) rows = BM; // OOB guard
  const bf16* gA = h + (size_t)(base + mloc0) * HID;
  BRegF br;
#if HAVE_TDM
  unsigned lds0 = (unsigned)(uintptr_t)&As[0][0];
  unsigned lds1 = (unsigned)(uintptr_t)&As[1][0];
  if (wave == 0) tdm_load_tile_bf16(lds0, gA, rows, HID);
  b_f32_load(br, W, OUTD, 0, n0);
  b_f32_store(Bs[0], br);
  if (wave == 0) __builtin_amdgcn_s_wait_tensorcnt(0);
  __syncthreads();
  int nk = HID / BK;
  for (int i = 0; i < nk; ++i) {
    int cur = i & 1;
    if (i + 1 < nk) {
      if (wave == 0)
        tdm_load_tile_bf16(cur ? lds0 : lds1, gA + (i + 1) * BK, rows, HID);
      b_f32_load(br, W, OUTD, (i + 1) * BK, n0);
    }
    mma_step(As[cur], Bs[cur], m_wave, n_wave, acc);
    if (i + 1 < nk) {
      b_f32_store(Bs[cur ^ 1], br);
      if (wave == 0) __builtin_amdgcn_s_wait_tensorcnt(0);
    }
    __syncthreads();
  }
#else
  ARegB ar;
  a_bf16_load(ar, h, HID, nullptr, base + mloc0, 0);
  b_f32_load(br, W, OUTD, 0, n0);
  a_bf16_store(As[0], ar);
  b_f32_store(Bs[0], br);
  __syncthreads();
  int nk = HID / BK;
  for (int i = 0; i < nk; ++i) {
    int cur = i & 1;
    if (i + 1 < nk) {
      a_bf16_load(ar, h, HID, nullptr, base + mloc0, (i + 1) * BK);
      b_f32_load(br, W, OUTD, (i + 1) * BK, n0);
    }
    mma_step(As[cur], Bs[cur], m_wave, n_wave, acc);
    if (i + 1 < nk) {
      a_bf16_store(As[cur ^ 1], ar);
      b_f32_store(Bs[cur ^ 1], br);
    }
    __syncthreads();
  }
#endif
  int l = threadIdx.x & 31;
  int mofs = (l >> 4) * 8;
#pragma unroll
  for (int nt = 0; nt < 2; ++nt) {
    int ncol = n0 + n_wave + nt * 16 + (l & 15);
    float bv = b2[e * OUTD + ncol];
#pragma unroll
    for (int mt = 0; mt < 4; ++mt)
#pragma unroll
      for (int r = 0; r < 8; ++r) {
        int mloc = mloc0 + m_wave + mt * 16 + mofs + r;
        if (mloc < cnt) {
          int row = base + mloc;
          float g = g_of_row[row];
          int slot = slot_of_row[row];
          slot_out[(size_t)slot * OUTD + ncol] = (bf16)((acc[mt][nt][r] + bv) * g);
        }
      }
  }
}

// ---------------------------------------------------------------------------
// 5) combine the two routed slots per token (deterministic, no atomics)
// ---------------------------------------------------------------------------
__global__ __launch_bounds__(256) void combine_kernel(
    const bf16* __restrict__ slot_out, float* __restrict__ out) {
  size_t i = (size_t)blockIdx.x * 256 + threadIdx.x;   // over B_TOK*OUTD
  size_t b = i >> 10;
  size_t o = i & 1023;
  out[i] = (float)slot_out[(b * 2) * (size_t)OUTD + o] +
           (float)slot_out[(b * 2 + 1) * (size_t)OUTD + o];
}

// ---------------------------------------------------------------------------
extern "C" void kernel_launch(void* const* d_in, const int* in_sizes, int n_in,
                              void* d_out, int out_size, void* d_ws, size_t ws_size,
                              hipStream_t stream) {
  const float* text  = (const float*)d_in[0];
  const float* image = (const float*)d_in[1];
  const float* noise = (const float*)d_in[2];
  const float* Wt    = (const float*)d_in[3];
  const float* bt    = (const float*)d_in[4];
  const float* Wi    = (const float*)d_in[5];
  const float* bi    = (const float*)d_in[6];
  const float* Wg    = (const float*)d_in[7];
  const float* bg    = (const float*)d_in[8];
  const float* W1    = (const float*)d_in[9];
  const float* b1    = (const float*)d_in[10];
  const float* W2    = (const float*)d_in[11];
  const float* b2    = (const float*)d_in[12];
  float* out = (float*)d_out;

  // workspace layout (~129 MiB): slot_out aliases combined (dead by GEMM2).
  char* ws = (char*)d_ws;
  bf16* combined = (bf16*)ws;                          // 64 MiB
  bf16* slot_out = (bf16*)ws;                          // alias, 64 MiB
  bf16* h        = (bf16*)(ws + ((size_t)64 << 20));   // 64 MiB
  char* p = ws + ((size_t)128 << 20);
  int*   choice_e    = (int*)p;   p += (size_t)NSLOT * 4;
  float* choice_g    = (float*)p; p += (size_t)NSLOT * 4;
  int*   tok_of_row  = (int*)p;   p += (size_t)NSLOT * 4;
  int*   slot_of_row = (int*)p;   p += (size_t)NSLOT * 4;
  float* g_of_row    = (float*)p; p += (size_t)NSLOT * 4;
  int*   counts  = (int*)p; p += 32;
  int*   bases   = (int*)p; p += 32;
  int*   cursors = (int*)p; p += 32;

  init_kernel<<<1, 32, 0, stream>>>(counts, cursors);
  proj_kernel<<<dim3(B_TOK / BM, 16), 256, 0, stream>>>(
      text, image, Wt, bt, Wi, bi, combined);
  gate_kernel<<<dim3(B_TOK / 8), 256, 0, stream>>>(
      combined, Wg, bg, noise, choice_e, choice_g, counts);
  scan_kernel<<<1, 1, 0, stream>>>(counts, bases);
  build_kernel<<<dim3(NSLOT / 256), 256, 0, stream>>>(
      choice_e, choice_g, bases, cursors, tok_of_row, slot_of_row, g_of_row);
  expert1_kernel<<<dim3(B_TOK / BM, HID / BN, NEXP), 256, 0, stream>>>(
      combined, W1, b1, counts, bases, tok_of_row, h);
  expert2_kernel<<<dim3(B_TOK / BM, OUTD / BN, NEXP), 256, 0, stream>>>(
      h, W2, b2, counts, bases, slot_of_row, g_of_row, slot_out);
  combine_kernel<<<dim3((B_TOK * OUTD) / 256), 256, 0, stream>>>(slot_out, out);
}
